// FAB_21285857919704
// MI455X (gfx1250) — compile-verified, matrix operand order checked by reading, not run on hardware
//
#include <hip/hip_runtime.h>
#include <stdint.h>

// ---------------- types & WMMA helpers ----------------
typedef __attribute__((ext_vector_type(16))) __bf16 v16bf;
typedef __attribute__((ext_vector_type(8)))  float  v8f;

union B16x16 { v16bf v; uint32_t d[8]; };

#define WMMA_BF16(A,B,C) __builtin_amdgcn_wmma_f32_16x16x32_bf16(false,(A),false,(B),(short)0,(C),false,false)

__device__ __forceinline__ v8f vzero() {
  v8f z = {0.f,0.f,0.f,0.f,0.f,0.f,0.f,0.f};
  return z;
}

// K offset of dword-pair p for this lane's half (CDNA5 16-bit A/B fragment layout):
// lanes 0-15: K = {0..7} then {16..23}; lanes 16-31: +8.
__device__ __forceinline__ int kofs(int p, int hi) {
  return hi * 8 + ((p < 4) ? 2 * p : 16 + 2 * (p - 4));
}

// Gather a 16x32 bf16 A (or 32x16 B) fragment from row-major memory (global or LDS):
// consecutive K pairs are contiguous -> 8 x b32 loads.
__device__ __forceinline__ v16bf gather_pairs(const __bf16* base, int hi) {
  B16x16 u;
#pragma unroll
  for (int p = 0; p < 8; ++p)
    u.d[p] = *(const uint32_t*)(base + kofs(p, hi));
  return u.v;
}

// One contiguous 32B fragment load per lane (fragment-major storage).
__device__ __forceinline__ v16bf ldsfrag(const __bf16* base, int lane) {
  return *((const v16bf*)base + lane);
}

// ---- CDNA5 async global->LDS staging (ASYNCcnt-tracked DMA-style copies) ----
// VDST = LDS byte address (low 32 bits of generic shared pointer), VADDR = 64-bit global addr.
__device__ __forceinline__ void async_cp16(uint32_t ldsaddr, const void* gaddr) {
  asm volatile("global_load_async_to_lds_b128 %0, %1, off"
               :: "v"(ldsaddr), "v"((uint64_t)(uintptr_t)gaddr) : "memory");
}
#define WAIT_ASYNC() asm volatile("s_wait_asynccnt 0x0" ::: "memory")

// all 256 threads cooperatively stage `bytes` contiguous bytes (mult of 16)
__device__ __forceinline__ void stage_async(const __bf16* src, const __bf16* dst, int bytes, int tid) {
  uint32_t lbase = (uint32_t)(uintptr_t)dst;
  const char* g = (const char*)src;
  for (int off = tid * 16; off < bytes; off += 256 * 16)
    async_cp16(lbase + off, g + off);
}

// ---------------- problem constants ----------------
#define C_    180
#define CP    192
#define NH_   6
#define HD_   30
#define NTOK  512
#define QKVN  540
#define HP    736
#define SCALE 0.18257418583505537f   // 30^-0.5

// ---------------- workspace layout (bytes) ----------------
static const size_t OFF_WQKV  = 0;
static const size_t OFF_WPROJ = 256u * 1024;
static const size_t OFF_WFC1  = 512u * 1024;
static const size_t OFF_WFC2  = 1024u * 1024;
static const size_t OFF_XRES  = 2u * 1024 * 1024;
static const size_t OFF_BUFA  = 26u * 1024 * 1024;   // xw -> ao -> h (aliased in time)
static const size_t OFF_BUFB  = 75u * 1024 * 1024;   // q_s -> y2
static const size_t OFF_BUFC  = 88u * 1024 * 1024;   // k_s
static const size_t OFF_BUFD  = 101u * 1024 * 1024;  // v_t

// ---------------- weight repack: W(outC,inC) f32 -> fragment-major bf16 ----------------
// B[k][n] = W[n][k]; frag index ntMajor ? nt*KT+kt : kt*NT+nt  (contiguous stage groups)
__global__ __launch_bounds__(256) void k_pack_b(const float* __restrict__ W, int outC, int inC,
                                                int Kpad, int Npad, int ntMajor,
                                                __bf16* __restrict__ dst) {
  int idx = blockIdx.x * 256 + threadIdx.x;
  int total = Kpad * Npad;
  if (idx >= total) return;
  int kk = idx / Npad, nn = idx % Npad;
  int kt = kk >> 5, nt = nn >> 4;
  int kin = kk & 31, nin = nn & 15;
  int hi = (kin >> 3) & 1;
  int L = nin + hi * 16;
  int k2 = kin - hi * 8;
  int e = (k2 < 8) ? k2 : (k2 - 8);
  int KT = Kpad >> 5, NT = Npad >> 4;
  int f = ntMajor ? (nt * KT + kt) : (kt * NT + nt);
  float v = (kk < inC && nn < outC) ? W[(size_t)nn * inC + kk] : 0.f;
  dst[(size_t)f * 512 + L * 16 + e] = (__bf16)v;
}

// ---------------- LN1 + cyclic shift + window partition -> xw bf16 (32768 x 192) ----------------
__global__ __launch_bounds__(256) void k_ln1(const float* __restrict__ x, const float* __restrict__ g,
                                             const float* __restrict__ b, __bf16* __restrict__ xw) {
  int lane = threadIdx.x & 31;
  int t = blockIdx.x * 8 + (threadIdx.x >> 5);
  int win = t >> 9, loc = t & 511;
  int gh = ((win >> 4) & 3) * 8 + (loc >> 6);
  int gw = ((win >> 2) & 3) * 8 + ((loc >> 3) & 7);
  int gd = (win & 3) * 8 + (loc & 7);
  int sh = (gh + 4) & 31, sw = (gw + 4) & 31, sd = (gd + 4) & 31;
  const float* row = x + (size_t)(((sh * 32) + sw) * 32 + sd) * C_;
  float s = 0.f, s2 = 0.f;
  for (int c = lane; c < C_; c += 32) { float v = row[c]; s += v; s2 += v * v; }
#pragma unroll
  for (int m = 1; m < 32; m <<= 1) { s += __shfl_xor(s, m, 32); s2 += __shfl_xor(s2, m, 32); }
  float mu = s * (1.f / C_);
  float inv = rsqrtf(s2 * (1.f / C_) - mu * mu + 1e-5f);
  __bf16* o = xw + (size_t)t * CP;
  for (int c = lane; c < CP; c += 32) {
    float v = (c < C_) ? ((row[c] - mu) * inv * g[c] + b[c]) : 0.f;
    o[c] = (__bf16)v;
  }
}

// ---------------- QKV GEMM: (32768 x 192) @ (192 x 544), async-staged B ----------------
__global__ __launch_bounds__(256) void k_qkv(const __bf16* __restrict__ xw, const __bf16* __restrict__ wp,
                                             const float* __restrict__ bias, __bf16* __restrict__ q_s,
                                             __bf16* __restrict__ k_s, __bf16* __restrict__ v_t) {
  __shared__ __align__(64) __bf16 bst[2][4 * 6 * 512];   // 2 x 24KB stage buffers
  int tid = threadIdx.x, lane = tid & 31, hi = lane >> 4, col = lane & 15;
  int mtile = blockIdx.x * 8 + (tid >> 5);
  int row = mtile * 16 + col;
  v16bf A[6];
#pragma unroll
  for (int kt = 0; kt < 6; ++kt) A[kt] = gather_pairs(xw + (size_t)row * CP + kt * 32, hi);

  stage_async(wp, bst[0], 4 * 6 * 1024, tid);            // group 0
  for (int g = 0; g < 9; ++g) {                          // ceil(34/4) nt-groups
    int nts = g * 4, nte = (nts + 4 < 34) ? nts + 4 : 34, buf = g & 1;
    WAIT_ASYNC(); __syncthreads();
    if (g < 8) {
      int s = (g + 1) * 4, e = (s + 4 < 34) ? s + 4 : 34;
      stage_async(wp + (size_t)s * 6 * 512, bst[buf ^ 1], (e - s) * 6 * 1024, tid);
    }
    for (int nt = nts; nt < nte; ++nt) {
      v16bf Bf[6];
#pragma unroll
      for (int kt = 0; kt < 6; ++kt) Bf[kt] = ldsfrag(&bst[buf][((nt - nts) * 6 + kt) * 512], lane);
      v8f acc = vzero();
#pragma unroll
      for (int kt = 0; kt < 6; ++kt) acc = WMMA_BF16(A[kt], Bf[kt], acc);
      int n = nt * 16 + col;
      if (n < QKVN) {
        float bb = bias[n];
        int which = n / 180, cc = n % 180, head = cc / 30, hd = cc % 30;
#pragma unroll
        for (int r = 0; r < 8; ++r) {
          int t = mtile * 16 + hi * 8 + r;
          int win = t >> 9, loc = t & 511, wh = win * NH_ + head;
          float v = acc[r] + bb;
          if (which == 0) {
            q_s[((size_t)(wh * NTOK + loc)) * 32 + hd] = (__bf16)(v * SCALE);
          } else if (which == 1) {
            k_s[((size_t)(wh * NTOK + loc)) * 32 + hd] = (__bf16)v;
            if (hd < 2) k_s[((size_t)(wh * NTOK + loc)) * 32 + 30 + hd] = (__bf16)0.f;
          } else {
            v_t[((size_t)(wh * 32 + hd)) * NTOK + loc] = (__bf16)v;
          }
        }
      }
    }
    __syncthreads();
  }
}

// ---------------- flash attention with async-staged K/V tiles ----------------
__global__ __launch_bounds__(256) void k_attn(const __bf16* __restrict__ q_s, const __bf16* __restrict__ k_s,
                                              const __bf16* __restrict__ v_t, const int* __restrict__ rpi,
                                              const float* __restrict__ rpb, const float* __restrict__ mask,
                                              __bf16* __restrict__ ao) {
  __shared__ __align__(64) __bf16 kst[2][32 * 32];   // K tile: [key][dim]
  __shared__ __align__(64) __bf16 vst[2][32 * 32];   // V tile: [dim][key]
  __shared__ __align__(64) __bf16 psm[8 * 16 * 32];  // per-wave P staging
  int tid = threadIdx.x, lane = tid & 31, w = tid >> 5, hi = lane >> 4, col = lane & 15;
  int wh = blockIdx.x;
  int win = wh / NH_, head = wh % NH_;
  int mtile = blockIdx.y * 8 + w;
  const __bf16* qb = q_s + (size_t)wh * NTOK * 32;
  const __bf16* kbp = k_s + (size_t)wh * NTOK * 32;
  const __bf16* vbp = v_t + (size_t)wh * 32 * NTOK;
  const float* mb = mask + (size_t)win * NTOK * NTOK;
  v16bf qa = gather_pairs(qb + (size_t)(mtile * 16 + col) * 32, hi);
  v8f O0 = vzero(), O1 = vzero();
  float mrow[8], lrow[8];
#pragma unroll
  for (int r = 0; r < 8; ++r) { mrow[r] = -3.0e38f; lrow[r] = 0.f; }
  __bf16* swm = psm + w * 512;

  // stage tile j: K contiguous (threads 0-127), V strided rows (threads 128-255)
  auto stage_kv = [&](int kb0, int buf) {
    if (tid < 128) {
      async_cp16((uint32_t)(uintptr_t)&kst[buf][0] + tid * 16,
                 (const char*)(kbp + (size_t)kb0 * 32) + tid * 16);
    } else {
      int c = tid - 128, dim = c >> 2, part = c & 3;
      async_cp16((uint32_t)(uintptr_t)&vst[buf][0] + dim * 64 + part * 16,
                 vbp + (size_t)dim * NTOK + kb0 + part * 8);
    }
  };

  stage_kv(0, 0);
  for (int j = 0; j < 16; ++j) {
    int buf = j & 1, kb0 = j * 32;
    WAIT_ASYNC(); __syncthreads();
    if (j < 15) stage_kv(kb0 + 32, buf ^ 1);

    v16bf kf0 = gather_pairs(&kst[buf][col * 32], hi);
    v16bf kf1 = gather_pairs(&kst[buf][(16 + col) * 32], hi);
    v8f S0 = WMMA_BF16(qa, kf0, vzero());
    v8f S1 = WMMA_BF16(qa, kf1, vzero());
#pragma unroll
    for (int r = 0; r < 8; ++r) {
      int row = mtile * 16 + hi * 8 + r;
      int key0 = kb0 + col, key1 = kb0 + 16 + col;
      S0[r] += rpb[(size_t)rpi[row * NTOK + key0] * NH_ + head] + mb[(size_t)row * NTOK + key0];
      S1[r] += rpb[(size_t)rpi[row * NTOK + key1] * NH_ + head] + mb[(size_t)row * NTOK + key1];
    }
#pragma unroll
    for (int r = 0; r < 8; ++r) {
      float t = fmaxf(S0[r], S1[r]);
#pragma unroll
      for (int mk = 1; mk < 16; mk <<= 1) t = fmaxf(t, __shfl_xor(t, mk, 32));
      float nm = fmaxf(mrow[r], t);
      float corr = __expf(mrow[r] - nm);
      mrow[r] = nm;
      float p0 = __expf(S0[r] - nm), p1 = __expf(S1[r] - nm);
      float ssum = p0 + p1;
#pragma unroll
      for (int mk = 1; mk < 16; mk <<= 1) ssum += __shfl_xor(ssum, mk, 32);
      lrow[r] = lrow[r] * corr + ssum;
      O0[r] *= corr; O1[r] *= corr;
      int rl = hi * 8 + r;
      swm[rl * 32 + col] = (__bf16)p0;
      swm[rl * 32 + 16 + col] = (__bf16)p1;
    }
    asm volatile("s_wait_dscnt 0x0" ::: "memory");   // wave-private LDS round-trip, DS in-order
    v16bf pa;
    {
      B16x16 u;
#pragma unroll
      for (int p = 0; p < 8; ++p)
        u.d[p] = *(const uint32_t*)&swm[col * 32 + kofs(p, hi)];
      pa = u.v;
    }
    v16bf vf0 = gather_pairs(&vst[buf][col * 32], hi);
    v16bf vf1 = gather_pairs(&vst[buf][(16 + col) * 32], hi);
    O0 = WMMA_BF16(pa, vf0, O0);
    O1 = WMMA_BF16(pa, vf1, O1);
    __syncthreads();
  }
  int dim0 = col, dim1 = 16 + col;
#pragma unroll
  for (int r = 0; r < 8; ++r) {
    int row = mtile * 16 + hi * 8 + r;
    float inv = 1.f / lrow[r];
    size_t base = ((size_t)(win * NTOK + row)) * CP + head * HD_;
    ao[base + dim0] = (__bf16)(O0[r] * inv);
    if (dim1 < HD_) ao[base + dim1] = (__bf16)(O1[r] * inv);
  }
}

// ---------------- proj GEMM + window reverse + roll(+4) + residual ----------------
__global__ __launch_bounds__(256) void k_proj(const __bf16* __restrict__ ao, const __bf16* __restrict__ wp,
                                              const float* __restrict__ bias, const float* __restrict__ x,
                                              float* __restrict__ xres) {
  __shared__ __align__(64) __bf16 bst[2][4 * 6 * 512];
  int tid = threadIdx.x, lane = tid & 31, hi = lane >> 4, col = lane & 15;
  int mtile = blockIdx.x * 8 + (tid >> 5);
  int row = mtile * 16 + col;
  v16bf A[6];
#pragma unroll
  for (int kt = 0; kt < 6; ++kt) A[kt] = gather_pairs(ao + (size_t)row * CP + kt * 32, hi);

  stage_async(wp, bst[0], 4 * 6 * 1024, tid);
  for (int g = 0; g < 3; ++g) {                           // 12 ntiles, groups of 4
    int nts = g * 4, buf = g & 1;
    WAIT_ASYNC(); __syncthreads();
    if (g < 2) stage_async(wp + (size_t)(nts + 4) * 6 * 512, bst[buf ^ 1], 4 * 6 * 1024, tid);
    for (int nt = nts; nt < nts + 4; ++nt) {
      v16bf Bf[6];
#pragma unroll
      for (int kt = 0; kt < 6; ++kt) Bf[kt] = ldsfrag(&bst[buf][((nt - nts) * 6 + kt) * 512], lane);
      v8f acc = vzero();
#pragma unroll
      for (int kt = 0; kt < 6; ++kt) acc = WMMA_BF16(A[kt], Bf[kt], acc);
      int n = nt * 16 + col;
      if (n < C_) {
        float bb = bias[n];
#pragma unroll
        for (int r = 0; r < 8; ++r) {
          int t = mtile * 16 + hi * 8 + r;
          int win = t >> 9, loc = t & 511;
          int gh = ((win >> 4) & 3) * 8 + (loc >> 6);
          int gw = ((win >> 2) & 3) * 8 + ((loc >> 3) & 7);
          int gd = (win & 3) * 8 + (loc & 7);
          int sh = (gh + 4) & 31, sw = (gw + 4) & 31, sd = (gd + 4) & 31;
          size_t ts = (size_t)(((sh * 32) + sw) * 32 + sd);
          xres[ts * C_ + n] = x[ts * C_ + n] + acc[r] + bb;
        }
      }
    }
    __syncthreads();
  }
}

// ---------------- LN2 -> y2 bf16 ----------------
__global__ __launch_bounds__(256) void k_ln2(const float* __restrict__ xr, const float* __restrict__ g,
                                             const float* __restrict__ b, __bf16* __restrict__ y2) {
  int lane = threadIdx.x & 31;
  int t = blockIdx.x * 8 + (threadIdx.x >> 5);
  const float* row = xr + (size_t)t * C_;
  float s = 0.f, s2 = 0.f;
  for (int c = lane; c < C_; c += 32) { float v = row[c]; s += v; s2 += v * v; }
#pragma unroll
  for (int m = 1; m < 32; m <<= 1) { s += __shfl_xor(s, m, 32); s2 += __shfl_xor(s2, m, 32); }
  float mu = s * (1.f / C_);
  float inv = rsqrtf(s2 * (1.f / C_) - mu * mu + 1e-5f);
  __bf16* o = y2 + (size_t)t * CP;
  for (int c = lane; c < CP; c += 32) {
    float v = (c < C_) ? ((row[c] - mu) * inv * g[c] + b[c]) : 0.f;
    o[c] = (__bf16)v;
  }
}

// ---------------- fc1 GEMM + exact GELU ----------------
__global__ __launch_bounds__(256) void k_fc1(const __bf16* __restrict__ y2, const __bf16* __restrict__ wp,
                                             const float* __restrict__ bias, __bf16* __restrict__ h) {
  __shared__ __align__(64) __bf16 bst[2][4 * 6 * 512];
  int tid = threadIdx.x, lane = tid & 31, hi = lane >> 4, col = lane & 15;
  int mtile = blockIdx.x * 8 + (tid >> 5);
  int row = mtile * 16 + col;
  v16bf A[6];
#pragma unroll
  for (int kt = 0; kt < 6; ++kt) A[kt] = gather_pairs(y2 + (size_t)row * CP + kt * 32, hi);

  stage_async(wp, bst[0], 4 * 6 * 1024, tid);
  for (int g = 0; g < 12; ++g) {                          // 45 ntiles, groups of 4 (last = 1)
    int nts = g * 4, nte = (nts + 4 < 45) ? nts + 4 : 45, buf = g & 1;
    WAIT_ASYNC(); __syncthreads();
    if (g < 11) {
      int s = (g + 1) * 4, e = (s + 4 < 45) ? s + 4 : 45;
      stage_async(wp + (size_t)s * 6 * 512, bst[buf ^ 1], (e - s) * 6 * 1024, tid);
    }
    for (int nt = nts; nt < nte; ++nt) {
      v16bf Bf[6];
#pragma unroll
      for (int kt = 0; kt < 6; ++kt) Bf[kt] = ldsfrag(&bst[buf][((nt - nts) * 6 + kt) * 512], lane);
      v8f acc = vzero();
#pragma unroll
      for (int kt = 0; kt < 6; ++kt) acc = WMMA_BF16(A[kt], Bf[kt], acc);
      int n = nt * 16 + col;
      float bb = bias[n];
#pragma unroll
      for (int r = 0; r < 8; ++r) {
        int t = mtile * 16 + hi * 8 + r;
        float v = acc[r] + bb;
        float ge = 0.5f * v * (1.f + erff(v * 0.70710678118654752f));
        h[(size_t)t * HP + n] = (__bf16)ge;
      }
    }
    __syncthreads();
  }
}

// ---------------- fc2 GEMM + bias + residual -> d_out f32 ----------------
__global__ __launch_bounds__(256) void k_fc2(const __bf16* __restrict__ h, const __bf16* __restrict__ wp,
                                             const float* __restrict__ bias, const float* __restrict__ xres,
                                             float* __restrict__ out) {
  __shared__ __align__(64) __bf16 bst[2][2 * 12 * 512];   // kt-major pack: groups of 2 kt
  int tid = threadIdx.x, lane = tid & 31, hi = lane >> 4, col = lane & 15;
  int mtile = blockIdx.x * 8 + (tid >> 5);
  int row = mtile * 16 + col;
  v8f acc[12];
#pragma unroll
  for (int nt = 0; nt < 12; ++nt) acc[nt] = vzero();

  stage_async(wp, bst[0], 2 * 12 * 1024, tid);
  for (int g = 0; g < 12; ++g) {                          // 23 ktiles, groups of 2 (last = 1)
    int kts = g * 2, kte = (kts + 2 < 23) ? kts + 2 : 23, buf = g & 1;
    WAIT_ASYNC(); __syncthreads();
    if (g < 11) {
      int s = (g + 1) * 2, e = (s + 2 < 23) ? s + 2 : 23;
      stage_async(wp + (size_t)s * 12 * 512, bst[buf ^ 1], (e - s) * 12 * 1024, tid);
    }
    for (int kt = kts; kt < kte; ++kt) {
      v16bf a = gather_pairs(h + (size_t)row * HP + kt * 32, hi);
#pragma unroll
      for (int nt = 0; nt < 12; ++nt)
        acc[nt] = WMMA_BF16(a, ldsfrag(&bst[buf][((kt - kts) * 12 + nt) * 512], lane), acc[nt]);
    }
    __syncthreads();
  }
#pragma unroll
  for (int nt = 0; nt < 12; ++nt) {
    int n = nt * 16 + col;
    if (n < C_) {
      float bb = bias[n];
#pragma unroll
      for (int r = 0; r < 8; ++r) {
        int t = mtile * 16 + hi * 8 + r;
        out[(size_t)t * C_ + n] = acc[nt][r] + bb + xres[(size_t)t * C_ + n];
      }
    }
  }
}

// ---------------- host launcher ----------------
extern "C" void kernel_launch(void* const* d_in, const int* in_sizes, int n_in,
                              void* d_out, int out_size, void* d_ws, size_t ws_size,
                              hipStream_t stream) {
  (void)in_sizes; (void)n_in; (void)out_size; (void)ws_size;
  const float* x     = (const float*)d_in[0];
  const int*   rpi   = (const int*)d_in[1];
  const float* amask = (const float*)d_in[2];
  const float* n1w   = (const float*)d_in[3];
  const float* n1b   = (const float*)d_in[4];
  const float* qkvw  = (const float*)d_in[5];
  const float* qkvb  = (const float*)d_in[6];
  const float* rpb   = (const float*)d_in[7];
  const float* projw = (const float*)d_in[8];
  const float* projb = (const float*)d_in[9];
  const float* n2w   = (const float*)d_in[10];
  const float* n2b   = (const float*)d_in[11];
  const float* fc1w  = (const float*)d_in[12];
  const float* fc1b  = (const float*)d_in[13];
  const float* fc2w  = (const float*)d_in[14];
  const float* fc2b  = (const float*)d_in[15];
  float* out = (float*)d_out;
  char* ws = (char*)d_ws;

  __bf16* Wqkv  = (__bf16*)(ws + OFF_WQKV);
  __bf16* Wproj = (__bf16*)(ws + OFF_WPROJ);
  __bf16* Wfc1  = (__bf16*)(ws + OFF_WFC1);
  __bf16* Wfc2  = (__bf16*)(ws + OFF_WFC2);
  float*  xres  = (float*)(ws + OFF_XRES);
  __bf16* xw    = (__bf16*)(ws + OFF_BUFA);
  __bf16* ao    = (__bf16*)(ws + OFF_BUFA);
  __bf16* hbuf  = (__bf16*)(ws + OFF_BUFA);
  __bf16* q_s   = (__bf16*)(ws + OFF_BUFB);
  __bf16* y2    = (__bf16*)(ws + OFF_BUFB);
  __bf16* k_s   = (__bf16*)(ws + OFF_BUFC);
  __bf16* v_t   = (__bf16*)(ws + OFF_BUFD);

  // repack weights (nt-major for k-inner GEMMs, kt-major for fc2)
  k_pack_b<<<(192 * 544 + 255) / 256, 256, 0, stream>>>(qkvw, 540, 180, 192, 544, 1, Wqkv);
  k_pack_b<<<(192 * 192 + 255) / 256, 256, 0, stream>>>(projw, 180, 180, 192, 192, 1, Wproj);
  k_pack_b<<<(192 * 720 + 255) / 256, 256, 0, stream>>>(fc1w, 720, 180, 192, 720, 1, Wfc1);
  k_pack_b<<<(736 * 192 + 255) / 256, 256, 0, stream>>>(fc2w, 180, 720, 736, 192, 0, Wfc2);

  k_ln1 <<<4096, 256, 0, stream>>>(x, n1w, n1b, xw);
  k_qkv <<<256, 256, 0, stream>>>(xw, Wqkv, qkvb, q_s, k_s, v_t);
  k_attn<<<dim3(384, 4), 256, 0, stream>>>(q_s, k_s, v_t, rpi, rpb, amask, ao);
  k_proj<<<256, 256, 0, stream>>>(ao, Wproj, projb, x, xres);
  k_ln2 <<<4096, 256, 0, stream>>>(xres, n2w, n2b, y2);
  k_fc1 <<<256, 256, 0, stream>>>(y2, Wfc1, fc1b, hbuf);
  k_fc2 <<<256, 256, 0, stream>>>(hbuf, Wfc2, fc2b, xres, out);
}
